// SelectionConv2d_18081812316977
// MI455X (gfx1250) — matrix operand order repaired and checked by reading, not compiled, not run on hardware
//
#include <hip/hip_runtime.h>

// ---- CDNA5 / gfx1250 WMMA types (wave32) ----
typedef __attribute__((ext_vector_type(16))) __bf16 v16bf;
typedef __attribute__((ext_vector_type(8)))  __bf16 v8bf;
typedef __attribute__((ext_vector_type(4)))  __bf16 v4bf;
typedef __attribute__((ext_vector_type(8)))  float  v8f;

#define C_IN   256      // K
#define O_OUT  256      // M
#define HW     4096     // pixels per sample
#define KB_N   8        // K-steps of 32
#define P_BLK  64       // pixels per block

// LDS staging: X tile [32 k x 64 p] as split bf16, [p][k] order.
// Row per p: 32 k * 2B = 64B data + 16B pad => 80B (16B-aligned b128 reads,
// odd bank stride to break the worst conflicts). hi and lo in separate halves.
#define LDS_ROW   80
#define LDS_HALF  (64 * LDS_ROW)   // 5120 B
#define LDS_BUF   (2 * LDS_HALF)   // hi + lo = 10240 B (x2 double buffer)

static __device__ __forceinline__ v16bf cat8(v8bf a, v8bf b) {
    return __builtin_shufflevector(a, b, 0,1,2,3,4,5,6,7,8,9,10,11,12,13,14,15);
}

// ---------------------------------------------------------------------------
// Pass 1: gather + split-convert weights into d_ws, pre-swizzled into the
// exact A-fragment VGPR layout of v_wmma_f32_16x16x32_bf16 (ISA 7.12.2):
//   lanes 0-15 : element e -> K = (e<8 ? e : e+8)
//   lanes 16-31: element e -> K = 8 + (e<8 ? e : e+8)
// Fragment f = ((n*16 + o_tile)*8 + kb): 32 lanes x 64B (hi 32B | lo 32B).
// Total: n*16*8 fragments * 2048B = 8 MB for n=32.
// ---------------------------------------------------------------------------
__global__ __launch_bounds__(256)
void selconv_prep_weights(const int*   __restrict__ classes,
                          const float* __restrict__ weight,
                          v8bf*        __restrict__ wsA,
                          int n_samples)
{
    const int g    = blockIdx.x * 8 + (threadIdx.x >> 5);  // one wave per fragment
    const int lane = threadIdx.x & 31;
    const int l16  = lane & 15;
    const int hh   = lane >> 4;

    const int kb = g & 7;
    const int ot = (g >> 3) & 15;
    const int n  = g >> 7;
    if (n >= n_samples) return;

    const int cls = classes[n];
    // lane holds weight row M = ot*16 + l16; contiguous in K.
    const float* Arow = weight + ((size_t)cls * O_OUT + ot * 16 + l16) * C_IN
                               + kb * 32 + hh * 8;

    v8bf h0, h1, l0, l1;
#pragma unroll
    for (int e = 0; e < 8; ++e) {
        const float a0 = Arow[e];        // K = kb*32 + hh*8 + e
        const float a1 = Arow[16 + e];   // K = kb*32 + hh*8 + 16 + e
        const __bf16 b0 = (__bf16)a0;
        const __bf16 b1 = (__bf16)a1;
        h0[e] = b0;  l0[e] = (__bf16)(a0 - (float)b0);
        h1[e] = b1;  l1[e] = (__bf16)(a1 - (float)b1);
    }
    v8bf* dst = wsA + ((size_t)g * 32 + lane) * 4;   // 64B per lane
    dst[0] = h0; dst[1] = h1; dst[2] = l0; dst[3] = l1;
}

// ---------------------------------------------------------------------------
// Pass 2: main GEMM. Block = 8 waves = 256 o x 64 p for one sample.
// Each wave: 2 o-tiles x 4 p-subtiles = 8 accumulators, 24 WMMAs / K-step.
// X tile converted to split-bf16 ONCE per block into LDS (double-buffered).
// grid = n_samples * (HW/64).
// ---------------------------------------------------------------------------
__global__ __launch_bounds__(256)
void selconv_main(const float* __restrict__ x,
                  const int*   __restrict__ classes,
                  const float* __restrict__ bias,
                  const v8bf*  __restrict__ wsA,
                  float*       __restrict__ out)
{
    __shared__ __align__(16) char lds[2 * LDS_BUF];   // 20 KB

    const int tid  = threadIdx.x;
    const int wave = tid >> 5;
    const int lane = tid & 31;
    const int l16  = lane & 15;
    const int hh   = lane >> 4;     // 0: lanes 0-15, 1: lanes 16-31

    const int p_blk = blockIdx.x & (HW / P_BLK - 1);  // 64 p-blocks
    const int n     = blockIdx.x >> 6;
    const int cls   = classes[n];
    const int p0    = p_blk * P_BLK;

    const float* Xn = x + (size_t)n * C_IN * HW + p0;

    // staging role: thread -> pixel column p_loc, 4-consecutive-k groups
    const int p_loc = tid & 63;
    const int k4    = (tid >> 6) * 4;   // 0,4,8,12

    v8f zero = {};
    v8f acc[2][4];
#pragma unroll
    for (int t = 0; t < 2; ++t)
#pragma unroll
        for (int s = 0; s < 4; ++s) acc[t][s] = zero;

#pragma unroll
    for (int kb = 0; kb < KB_N; ++kb) {
        char* buf = lds + (kb & 1) * LDS_BUF;

        // ---- stage X tile (split-bf16, converted once per block) ----
        // B-fragment layout per ISA 7.12.2: lane l, element e ->
        //   K = (l>=16 ? 8 : 0) + (e<8 ? e : e+8), N = l%16  => [p][k] rows in LDS.
#pragma unroll
        for (int j = 0; j < 2; ++j) {
            const int k = k4 + 16 * j;
            v4bf h, l;
#pragma unroll
            for (int i = 0; i < 4; ++i) {
                const float v = Xn[(kb * 32 + k + i) * HW + p_loc];
                const __bf16 hb = (__bf16)v;
                h[i] = hb;
                l[i] = (__bf16)(v - (float)hb);
            }
            *(v4bf*)(buf + p_loc * LDS_ROW + k * 2) = h;                 // ds_store_b64
            *(v4bf*)(buf + LDS_HALF + p_loc * LDS_ROW + k * 2) = l;      // ds_store_b64
        }

        // ---- A fragments: pre-swizzled bf16 hi/lo straight from d_ws ----
        v16bf a_hi[2], a_lo[2];
#pragma unroll
        for (int t = 0; t < 2; ++t) {
            const int ot = wave * 2 + t;
            const v8bf* Af =
                wsA + ((size_t)(((n * 16 + ot) * 8) + kb) * 32 + lane) * 4;
            a_hi[t] = cat8(Af[0], Af[1]);   // 2x global_load_b128
            a_lo[t] = cat8(Af[2], Af[3]);   // 2x global_load_b128
        }

        __syncthreads();   // one barrier per K-step (double-buffered LDS)

        // ---- consume: 4 B subtiles x 2 o-tiles x 3 split-WMMAs ----
#pragma unroll
        for (int s = 0; s < 4; ++s) {
            const char* rb = buf + (s * 16 + l16) * LDS_ROW + hh * 16;
            const v16bf b_hi = cat8(*(const v8bf*)rb,
                                    *(const v8bf*)(rb + 32));            // 2x ds_load_b128
            const v16bf b_lo = cat8(*(const v8bf*)(rb + LDS_HALF),
                                    *(const v8bf*)(rb + LDS_HALF + 32)); // 2x ds_load_b128
#pragma unroll
            for (int t = 0; t < 2; ++t) {
                acc[t][s] = __builtin_amdgcn_wmma_f32_16x16x32_bf16(
                    false, a_lo[t], false, b_hi, (short)0, acc[t][s], false, false);
                acc[t][s] = __builtin_amdgcn_wmma_f32_16x16x32_bf16(
                    false, a_hi[t], false, b_lo, (short)0, acc[t][s], false, false);
                acc[t][s] = __builtin_amdgcn_wmma_f32_16x16x32_bf16(
                    false, a_hi[t], false, b_hi, (short)0, acc[t][s], false, false);
            }
        }
    }

    // ---- epilogue: bias fused, D layout VGPR r -> M = r + hh*8, N = l16 ----
    float*       On = out  + (size_t)n   * O_OUT * HW;
    const float* Bn = bias + (size_t)cls * O_OUT;
#pragma unroll
    for (int t = 0; t < 2; ++t) {
#pragma unroll
        for (int r = 0; r < 8; ++r) {
            const int   o  = wave * 32 + t * 16 + hh * 8 + r;
            const float bv = Bn[o];
#pragma unroll
            for (int s = 0; s < 4; ++s) {
                On[(size_t)o * HW + p0 + s * 16 + l16] = acc[t][s][r] + bv;
            }
        }
    }
}

// ---------------------------------------------------------------------------
extern "C" void kernel_launch(void* const* d_in, const int* in_sizes, int n_in,
                              void* d_out, int out_size, void* d_ws, size_t ws_size,
                              hipStream_t stream) {
    const float* x       = (const float*)d_in[0]; // [32,256,64,64]
    const int*   classes = (const int*)  d_in[1]; // [32]
    const float* weight  = (const float*)d_in[2]; // [100,256,256]
    const float* bias    = (const float*)d_in[3]; // [100,256]
    float*       out     = (float*)d_out;         // [32,256,64,64]

    const int n_samples = in_sizes[1];            // 32
    v8bf* wsA = (v8bf*)d_ws;                      // needs n*16*8*2048 B = 8 MB

    // Pass 1: gather + split + swizzle weights (one wave per 16x32 fragment).
    const int frags      = n_samples * 16 * KB_N;         // 4096
    const int prep_grid  = (frags + 7) / 8;               // 8 waves / block
    selconv_prep_weights<<<prep_grid, 256, 0, stream>>>(classes, weight, wsA,
                                                        n_samples);

    // Pass 2: main GEMM, grid = n * 64 p-blocks.
    const int main_grid = n_samples * (HW / P_BLK);       // 2048
    selconv_main<<<main_grid, 256, 0, stream>>>(x, classes, bias, wsA, out);
}
